// DogeCDMoE_78778290144068
// MI455X (gfx1250) — matrix operand order repaired and checked by reading, not compiled, not run on hardware
//
#include <hip/hip_runtime.h>
#include <hip/hip_bf16.h>

// Problem constants (from the reference)
#define BATCH     2
#define TSEQ      2048
#define NTOK      (BATCH * TSEQ)   // 4096 tokens
#define HDIM      1024             // H
#define IDIM      4096             // I
#define HEADS     4
#define RET       128
#define HALF_RET  (RET / 2)        // 64
#define QD        (HEADS * RET)    // 512
#define NKEYS     64
#define KSEL      8

typedef __attribute__((ext_vector_type(8)))  float  v8f;
typedef __attribute__((ext_vector_type(16))) __bf16 v16bf;
typedef __attribute__((ext_vector_type(8)))  __bf16 v8bf;
typedef __attribute__((ext_vector_type(4)))  int    v4i;

__device__ __forceinline__ unsigned short f2bf(float f) {
    unsigned int u = __float_as_uint(f);
    unsigned int r = u + 0x7FFFu + ((u >> 16) & 1u);   // round-to-nearest-even
    return (unsigned short)(r >> 16);
}

// ---------------------------------------------------------------------------
// fp32 -> bf16 conversion, 4 elements / thread (n always divisible by 4 here)
// ---------------------------------------------------------------------------
__global__ __launch_bounds__(256)
void f32_to_bf16_kernel(const float4* __restrict__ src,
                        ushort4* __restrict__ dst, int n4) {
    int i = blockIdx.x * 256 + threadIdx.x;
    if (i < n4) {
        float4 f = src[i];
        ushort4 o;
        o.x = f2bf(f.x); o.y = f2bf(f.y); o.z = f2bf(f.z); o.w = f2bf(f.w);
        dst[i] = o;
    }
}

// ---------------------------------------------------------------------------
// NT GEMM: C[M,N] = A[M,K] * B[N,K]^T, A/B bf16 row-major (K contiguous),
// fp32 accumulate via V_WMMA_F32_16X16X32_BF16.
// Wave tile 32x64 (2 A frags x 4 B frags -> 8 WMMAs, 12 b128 loads / k-step).
// Block = 256 threads = 8 waves arranged 2(M) x 4(N) -> 64x256 block tile.
// Requires: M % 64 == 0, N % 256 == 0, K % 32 == 0 (true for all three uses).
// ---------------------------------------------------------------------------
template<int SILU, int OUT_BF16>
__global__ __launch_bounds__(256)
void gemm_bf16_nt(const unsigned short* __restrict__ Aus,
                  const unsigned short* __restrict__ Bus,
                  float* __restrict__ Cf,
                  unsigned short* __restrict__ Cb,
                  int M, int N, int Kd) {
    const __bf16* A = (const __bf16*)Aus;
    const __bf16* B = (const __bf16*)Bus;

    const int lane = threadIdx.x & 31;
    const int wave = threadIdx.x >> 5;
    const int wm   = wave & 1;       // 0..1  (M sub-tile of 32)
    const int wn   = wave >> 1;      // 0..3  (N sub-tile of 64)
    const int m0   = blockIdx.y * 64  + wm * 32;
    const int n0   = blockIdx.x * 256 + wn * 64;
    const int l16  = lane & 15;
    const int hi   = lane >> 4;      // K-half selector per ISA fragment layout

    const __bf16* arow[2];
    const __bf16* brow[4];
#pragma unroll
    for (int mi = 0; mi < 2; ++mi)
        arow[mi] = A + (size_t)(m0 + mi * 16 + l16) * Kd;
#pragma unroll
    for (int ni = 0; ni < 4; ++ni)
        brow[ni] = B + (size_t)(n0 + ni * 16 + l16) * Kd;

    const int aoff = hi * 8;         // A lane K base within 32-wide slab
    const int boff = hi * 8;         // B lane K base within 32-wide slab

    v8f acc[2][4];
#pragma unroll
    for (int mi = 0; mi < 2; ++mi)
#pragma unroll
        for (int ni = 0; ni < 4; ++ni)
#pragma unroll
            for (int i = 0; i < 8; ++i) acc[mi][ni][i] = 0.0f;

    for (int k0 = 0; k0 < Kd; k0 += 32) {
        union { v16bf v; v8bf h[2]; } a[2], b[4];
#pragma unroll
        for (int mi = 0; mi < 2; ++mi) {
            a[mi].h[0] = *(const v8bf*)(arow[mi] + k0 + aoff);
            a[mi].h[1] = *(const v8bf*)(arow[mi] + k0 + aoff + 16);
        }
#pragma unroll
        for (int ni = 0; ni < 4; ++ni) {
            b[ni].h[0] = *(const v8bf*)(brow[ni] + k0 + boff);
            b[ni].h[1] = *(const v8bf*)(brow[ni] + k0 + boff + 16);
        }

        __builtin_prefetch(arow[0] + k0 + 512, 0, 0);   // global_prefetch_b8
        __builtin_prefetch(brow[0] + k0 + 512, 0, 0);
        __builtin_prefetch(brow[2] + k0 + 512, 0, 0);

#pragma unroll
        for (int mi = 0; mi < 2; ++mi)
#pragma unroll
            for (int ni = 0; ni < 4; ++ni)
                acc[mi][ni] = __builtin_amdgcn_wmma_f32_16x16x32_bf16(
                    false, a[mi].v, false, b[ni].v, (short)0, acc[mi][ni],
                    false, false);
    }

    // C layout: VGPR r -> row (tile_m + r + 8*hi), col (tile_n + l16)
#pragma unroll
    for (int mi = 0; mi < 2; ++mi) {
#pragma unroll
        for (int ni = 0; ni < 4; ++ni) {
#pragma unroll
            for (int r = 0; r < 8; ++r) {
                float v = acc[mi][ni][r];
                if (SILU) v = v / (1.0f + __expf(-v));
                size_t off = (size_t)(m0 + mi * 16 + r + 8 * hi) * N
                           + (size_t)(n0 + ni * 16 + l16);
                if (OUT_BF16) Cb[off] = f2bf(v);
                else          Cf[off] = v;
            }
        }
    }
}

// ---------------------------------------------------------------------------
// Fused product-key retrieval + expert compute. One block (256 thr) per token.
//   sim[p,h,k] = q[token, p*256+h*64 : +64] . keys[h,k,p,:]
//   top8 per (p,h); Cartesian 8x8 -> top8 per h; softmax; expert gather.
//   out[token] += sum_{h,k} silu(hs.de * prob) * ue
// ---------------------------------------------------------------------------
__global__ __launch_bounds__(256)
void retrieval_experts_kernel(const float* __restrict__ hs,
                              const float* __restrict__ qbuf,
                              const float* __restrict__ keys,
                              const float* __restrict__ down_embed,
                              const float* __restrict__ up_embed,
                              float* __restrict__ out) {
    __shared__ float s_hs[HDIM];
    __shared__ float s_sim[2 * HEADS * NKEYS];   // 512, row = p*4+h
    __shared__ float s_tv[8][KSEL];
    __shared__ int   s_ti[8][KSEL];
    __shared__ float s_prob[HEADS][KSEL];
    __shared__ int   s_eidx[HEADS][KSEL];
    __shared__ float s_w[HEADS][KSEL];

    const int tid   = threadIdx.x;
    const int token = blockIdx.x;

    // stage token activations in LDS (async DMA path when available)
#if __has_builtin(__builtin_amdgcn_global_load_async_to_lds_b128) && \
    __has_builtin(__builtin_amdgcn_s_wait_asynccnt)
    {
        const float* g = hs + (size_t)token * HDIM + tid * 4;   // 256*16B = 4KB
        __builtin_amdgcn_global_load_async_to_lds_b128(
            (__attribute__((address_space(1))) v4i*)g,
            (__attribute__((address_space(3))) v4i*)(&s_hs[tid * 4]),
            0, 0);
        __builtin_amdgcn_s_wait_asynccnt(0);
    }
#else
    for (int i = tid; i < HDIM; i += 256)
        s_hs[i] = hs[(size_t)token * HDIM + i];
#endif

    // similarities: 512 dots of length 64
    for (int r = tid; r < 512; r += 256) {
        int p = r >> 8, h = (r >> 6) & 3, kk = r & 63;
        const float* qv = qbuf + (size_t)token * QD + p * 256 + h * 64;
        const float* kv = keys + (((size_t)(h * NKEYS + kk)) * 2 + p) * HALF_RET;
        float s = 0.f;
#pragma unroll 8
        for (int j = 0; j < HALF_RET; ++j) s += qv[j] * kv[j];
        s_sim[r] = s;
    }
    __syncthreads();

    // per-(p,h) top-8 (8 independent rows of 64; destructive scan in LDS)
    if (tid < 8) {
        float* row = s_sim + tid * NKEYS;
        for (int sel = 0; sel < KSEL; ++sel) {
            float best = -3.0e38f; int bi = 0;
            for (int j = 0; j < NKEYS; ++j) {
                float v = row[j];
                if (v > best) { best = v; bi = j; }
            }
            s_tv[tid][sel] = best;
            s_ti[tid][sel] = bi;
            row[bi] = -3.0e38f;
        }
    }
    __syncthreads();

    // Cartesian combine + top-8 + softmax, one thread per head
    if (tid < HEADS) {
        const int h = tid;
        const float* sx = s_tv[0 * HEADS + h]; const int* ixv = s_ti[0 * HEADS + h];
        const float* sy = s_tv[1 * HEADS + h]; const int* iyv = s_ti[1 * HEADS + h];
        unsigned long long used = 0ULL;
        float cs[KSEL]; int ci[KSEL];
        for (int sel = 0; sel < KSEL; ++sel) {
            float best = -3.0e38f; int bc = 0;
            for (int c = 0; c < 64; ++c) {
                if (used & (1ULL << c)) continue;
                float v = sx[c >> 3] + sy[c & 7];
                if (v > best) { best = v; bc = c; }
            }
            used |= (1ULL << bc);
            cs[sel] = best;
            ci[sel] = ixv[bc >> 3] * NKEYS + iyv[bc & 7];
        }
        float m = cs[0];
        for (int s = 1; s < KSEL; ++s) m = fmaxf(m, cs[s]);
        float sum = 0.f, e[KSEL];
        for (int s = 0; s < KSEL; ++s) { e[s] = __expf(cs[s] - m); sum += e[s]; }
        float inv = 1.0f / sum;
        for (int s = 0; s < KSEL; ++s) {
            s_prob[h][s] = e[s] * inv;
            s_eidx[h][s] = ci[s];
        }
    }
    __syncthreads();

    // 32 expert dot products (hs . down_embed[idx]); one wave per dot
    const int lane = tid & 31, wave = tid >> 5;
    for (int it = 0; it < 4; ++it) {
        int pair = it * 8 + wave;            // 0..31
        int h = pair >> 3, k = pair & 7;
        const float* de = down_embed + (size_t)s_eidx[h][k] * HDIM;
        float part = 0.f;
        for (int j = lane; j < HDIM; j += 32) part += s_hs[j] * de[j];
#pragma unroll
        for (int off = 16; off > 0; off >>= 1)
            part += __shfl_down(part, off, 32);
        if (lane == 0) {
            float x = part * s_prob[h][k];
            s_w[h][k] = x / (1.0f + __expf(-x));   // silu
        }
    }
    __syncthreads();

    // out[token] += sum_{h,k} w * up_embed[idx]
    for (int d = tid; d < HDIM; d += 256) {
        float acc = out[(size_t)token * HDIM + d];
#pragma unroll
        for (int pair = 0; pair < 32; ++pair) {
            int h = pair >> 3, k = pair & 7;
            acc += s_w[h][k] * up_embed[(size_t)s_eidx[h][k] * HDIM + d];
        }
        out[(size_t)token * HDIM + d] = acc;
    }
}

// ---------------------------------------------------------------------------
// Workspace layout (bytes)
// ---------------------------------------------------------------------------
static const size_t OFF_HS    = 0;                                  // 8 MB  bf16 hs
static const size_t OFF_WUP   = OFF_HS   + (size_t)NTOK * HDIM * 2; // 8 MB  bf16 W_up
static const size_t OFF_WDOWN = OFF_WUP  + (size_t)IDIM * HDIM * 2; // 8 MB  bf16 W_down
static const size_t OFF_WQ    = OFF_WDOWN+ (size_t)HDIM * IDIM * 2; // 1 MB  bf16 W_q
static const size_t OFF_ACT   = OFF_WQ   + (size_t)QD   * HDIM * 2; // 32 MB bf16 act
static const size_t OFF_Q     = OFF_ACT  + (size_t)NTOK * IDIM * 2; // 8 MB  f32 q

extern "C" void kernel_launch(void* const* d_in, const int* in_sizes, int n_in,
                              void* d_out, int out_size, void* d_ws, size_t ws_size,
                              hipStream_t stream) {
    const float* hs         = (const float*)d_in[0];
    const float* W_up       = (const float*)d_in[1];
    const float* W_down     = (const float*)d_in[2];
    const float* W_q        = (const float*)d_in[3];
    const float* keys       = (const float*)d_in[4];
    const float* down_embed = (const float*)d_in[5];
    const float* up_embed   = (const float*)d_in[6];
    float* out = (float*)d_out;

    char* ws = (char*)d_ws;
    unsigned short* hs_bf    = (unsigned short*)(ws + OFF_HS);
    unsigned short* wup_bf   = (unsigned short*)(ws + OFF_WUP);
    unsigned short* wdown_bf = (unsigned short*)(ws + OFF_WDOWN);
    unsigned short* wq_bf    = (unsigned short*)(ws + OFF_WQ);
    unsigned short* act_bf   = (unsigned short*)(ws + OFF_ACT);
    float*          qbuf     = (float*)(ws + OFF_Q);

    // 1) fp32 -> bf16 conversions (vectorized x4)
    {
        int n4;
        n4 = NTOK * HDIM / 4;
        f32_to_bf16_kernel<<<(n4 + 255) / 256, 256, 0, stream>>>(
            (const float4*)hs, (ushort4*)hs_bf, n4);
        n4 = IDIM * HDIM / 4;
        f32_to_bf16_kernel<<<(n4 + 255) / 256, 256, 0, stream>>>(
            (const float4*)W_up, (ushort4*)wup_bf, n4);
        n4 = HDIM * IDIM / 4;
        f32_to_bf16_kernel<<<(n4 + 255) / 256, 256, 0, stream>>>(
            (const float4*)W_down, (ushort4*)wdown_bf, n4);
        n4 = QD * HDIM / 4;
        f32_to_bf16_kernel<<<(n4 + 255) / 256, 256, 0, stream>>>(
            (const float4*)W_q, (ushort4*)wq_bf, n4);
    }

    dim3 blk(256);
    // 2) act = silu(hs @ W_up^T), bf16 out : M=4096 N=4096 K=1024
    gemm_bf16_nt<1, 1><<<dim3(IDIM / 256, NTOK / 64), blk, 0, stream>>>(
        hs_bf, wup_bf, nullptr, act_bf, NTOK, IDIM, HDIM);
    // 3) q = hs @ W_q^T, f32 out : M=4096 N=512 K=1024
    gemm_bf16_nt<0, 0><<<dim3(QD / 256, NTOK / 64), blk, 0, stream>>>(
        hs_bf, wq_bf, qbuf, nullptr, NTOK, QD, HDIM);
    // 4) dense = act @ W_down^T -> d_out, f32 : M=4096 N=1024 K=4096
    gemm_bf16_nt<0, 0><<<dim3(HDIM / 256, NTOK / 64), blk, 0, stream>>>(
        act_bf, wdown_bf, out, nullptr, NTOK, HDIM, IDIM);

    // 5) retrieval + experts, adds onto d_out
    retrieval_experts_kernel<<<NTOK, 256, 0, stream>>>(
        hs, qbuf, keys, down_embed, up_embed, out);
}
